// MesoNet_4088808866418
// MI455X (gfx1250) — matrix operand order repaired and checked by reading, not compiled, not run on hardware
//
#include <hip/hip_runtime.h>
#include <math.h>

typedef __attribute__((ext_vector_type(16))) __bf16 v16bf;
typedef __attribute__((ext_vector_type(8)))  float  v8f;

#define DEVI __device__ __forceinline__

namespace {

constexpr int kNA   = 120000;
constexpr int kG    = 30000;
constexpr int kNINC = 240000;
constexpr int kE    = 120000;
constexpr int kD    = 160;
constexpr int kHALF = 80;

DEVI float sigmoidf_(float x){ return 1.f/(1.f+__expf(-x)); }

DEVI float waveAllReduce(float s){
#pragma unroll
  for (int o = 16; o; o >>= 1) s += __shfl_xor(s, o, 32);
  return s;
}

// float atomic max via signed-max / unsigned-min trick (buffer init to -inf)
DEVI void atomicMaxF(float* addr, float v){
  if (v >= 0.f) atomicMax((int*)addr, __float_as_int(v));
  else          atomicMin((unsigned int*)addr, __float_as_uint(v));
}

// ---- WMMA fragment builders (layouts per CDNA5 ISA 7.12.2, 16-bit, wave32) ----
// A 16x32 (MxK): lane&15 = row; lane half selects kh = 0/8.
//   a[0..7]  = K = kh .. kh+7      (contiguous)
//   a[8..15] = K = 16+kh .. 16+kh+7 (contiguous)
// Fully in-bounds vectorized load: 4 x float4 from the lane's row.
DEVI v16bf loadA_vec(const float* __restrict__ rowp, int lane){
  const int kh = (lane >> 4) * 8;
  const float4 x0 = *(const float4*)(rowp + kh);
  const float4 x1 = *(const float4*)(rowp + kh + 4);
  const float4 x2 = *(const float4*)(rowp + 16 + kh);
  const float4 x3 = *(const float4*)(rowp + 16 + kh + 4);
  v16bf a;
  a[0]=(__bf16)x0.x;  a[1]=(__bf16)x0.y;  a[2]=(__bf16)x0.z;  a[3]=(__bf16)x0.w;
  a[4]=(__bf16)x1.x;  a[5]=(__bf16)x1.y;  a[6]=(__bf16)x1.z;  a[7]=(__bf16)x1.w;
  a[8]=(__bf16)x2.x;  a[9]=(__bf16)x2.y;  a[10]=(__bf16)x2.z; a[11]=(__bf16)x2.w;
  a[12]=(__bf16)x3.x; a[13]=(__bf16)x3.y; a[14]=(__bf16)x3.z; a[15]=(__bf16)x3.w;
  return a;
}
// B 32x16 (KxN): lane&15 = column; lane half selects kh = 0/8.
//   b[0..7] = K = kh .. kh+7, b[8..15] = K = 16+kh .. 16+kh+7 (same as A)
DEVI v16bf loadB_vec(const float* __restrict__ colp, int lane){
  const int kh = (lane >> 4) * 8;
  const float4 x0 = *(const float4*)(colp + kh);
  const float4 x1 = *(const float4*)(colp + kh + 4);
  const float4 x2 = *(const float4*)(colp + 16 + kh);
  const float4 x3 = *(const float4*)(colp + 16 + kh + 4);
  v16bf b;
  b[0]=(__bf16)x0.x;  b[1]=(__bf16)x0.y;  b[2]=(__bf16)x0.z;  b[3]=(__bf16)x0.w;
  b[4]=(__bf16)x1.x;  b[5]=(__bf16)x1.y;  b[6]=(__bf16)x1.z;  b[7]=(__bf16)x1.w;
  b[8]=(__bf16)x2.x;  b[9]=(__bf16)x2.y;  b[10]=(__bf16)x2.z; b[11]=(__bf16)x2.w;
  b[12]=(__bf16)x3.x; b[13]=(__bf16)x3.y; b[14]=(__bf16)x3.z; b[15]=(__bf16)x3.w;
  return b;
}
// Guarded scalar builders for the single ragged-K tail tile (K=40 layer only).
// dk is the in-tile K offset (0..31); f(dk) must zero-fill out-of-range.
template <typename F>
DEVI v16bf makeAFragG(F f, int lane){
  const int kh = (lane >> 4) * 8;
  v16bf a;
#pragma unroll
  for (int j = 0; j < 8; ++j){ a[j] = (__bf16)f(kh + j); a[8+j] = (__bf16)f(16 + kh + j); }
  return a;
}
template <typename F>
DEVI v16bf makeBFragG(F f, int lane){
  const int kh = (lane >> 4) * 8;
  v16bf b;
#pragma unroll
  for (int j = 0; j < 8; ++j){ b[j] = (__bf16)f(kh + j); b[8+j] = (__bf16)f(16 + kh + j); }
  return b;
}

// Generic C = act(A @ W^T + bias + add). A is fp32 [M,K] (optionally split into
// A0 cols [0,splitA) and A1 cols [splitA,K), splitA a multiple of 32);
// W is fp32 [N,K] row-major. One wave per 16x16 output tile.
__global__ void gemm_bf16_kernel(const float* __restrict__ A0, const float* __restrict__ A1,
                                 int lda0, int lda1, int splitA,
                                 const float* __restrict__ W, int ldw,
                                 const float* __restrict__ bias,
                                 const float* __restrict__ add, int ldadd,
                                 float* __restrict__ C, int ldc,
                                 int M, int N, int K, int relu)
{
  const int lane = threadIdx.x & 31;
  const int wave = (int)((blockIdx.x * blockDim.x + threadIdx.x) >> 5);
  const int tn = N >> 4;
  const int tiles = (M >> 4) * tn;
  if (wave >= tiles) return;
  const int bm = (wave / tn) << 4;
  const int bn = (wave % tn) << 4;
  const int arow = bm + (lane & 15);
  const int bcol = bn + (lane & 15);
  const float* __restrict__ arow0 = A0 + (size_t)arow * lda0;
  const float* __restrict__ arow1 = A1 + (size_t)arow * lda1;
  const float* __restrict__ wrow  = W  + (size_t)bcol * ldw;
  v8f acc = {0.f,0.f,0.f,0.f,0.f,0.f,0.f,0.f};
  for (int k0 = 0; k0 < K; k0 += 32){
    // tile-uniform segment select (splitA is a multiple of 32)
    const float* aseg = (k0 < splitA) ? (arow0 + k0) : (arow1 + (k0 - splitA));
    v16bf a, b;
    if (k0 + 32 <= K){
      a = loadA_vec(aseg, lane);
      b = loadB_vec(wrow + k0, lane);
      if (k0 + 64 <= K){
        __builtin_prefetch((k0 + 32 < splitA) ? (arow0 + k0 + 32) : (arow1 + (k0 + 32 - splitA)), 0, 1);
        __builtin_prefetch(wrow + k0 + 32, 0, 1);
      }
    } else {
      a = makeAFragG([&](int dk)->float{ return (k0 + dk < K) ? aseg[dk] : 0.f; }, lane);
      b = makeBFragG([&](int dk)->float{ return (k0 + dk < K) ? wrow[k0 + dk] : 0.f; }, lane);
    }
    acc = __builtin_amdgcn_wmma_f32_16x16x32_bf16(false, a, false, b, (short)0, acc, false, false);
  }
#pragma unroll
  for (int rr = 0; rr < 8; ++rr){
    int m = bm + rr + ((lane >> 4) << 3);
    float v = acc[rr];
    if (bias) v += bias[bcol];
    if (add)  v += add[(size_t)m*ldadd + bcol];
    if (relu) v = v > 0.f ? v : 0.f;
    C[(size_t)m*ldc + bcol] = v;
  }
}

// Fused Set2Set LSTM step: gates = [h|r|h] @ [Wih|Whh]^T + b, then LSTM cell.
// One wave computes the 4 gate tiles (i,f,g,o) for a 16x16 (row, hidden-col) tile.
// K = 480 exactly (15 full tiles); segment boundaries at 160/320 are multiples of 32.
__global__ void lstm_gemm_cell_kernel(const float* __restrict__ h_in, const float* __restrict__ r,
                                      const float* __restrict__ Wih, const float* __restrict__ Whh,
                                      const float* __restrict__ bih, const float* __restrict__ bhh,
                                      float* __restrict__ c, float* __restrict__ h_out, int R)
{
  const int lane = threadIdx.x & 31;
  const int wave = (int)((blockIdx.x * blockDim.x + threadIdx.x) >> 5);
  const int tn = kD >> 4;             // 10
  const int tiles = (R >> 4) * tn;
  if (wave >= tiles) return;
  const int bm = (wave / tn) << 4;
  const int bn = (wave % tn) << 4;
  const int arow = bm + (lane & 15);
  const int ncol = bn + (lane & 15);
  const float* __restrict__ hrow = h_in + (size_t)arow * kD;
  const float* __restrict__ rrow = r    + (size_t)arow * kD;
  const float* __restrict__ wI = Wih + (size_t)(0*kD + ncol) * 320;
  const float* __restrict__ wF = Wih + (size_t)(1*kD + ncol) * 320;
  const float* __restrict__ wG = Wih + (size_t)(2*kD + ncol) * 320;
  const float* __restrict__ wO = Wih + (size_t)(3*kD + ncol) * 320;
  const float* __restrict__ uI = Whh + (size_t)(0*kD + ncol) * kD;
  const float* __restrict__ uF = Whh + (size_t)(1*kD + ncol) * kD;
  const float* __restrict__ uG = Whh + (size_t)(2*kD + ncol) * kD;
  const float* __restrict__ uO = Whh + (size_t)(3*kD + ncol) * kD;
  v8f accI = {0.f,0.f,0.f,0.f,0.f,0.f,0.f,0.f};
  v8f accF = accI, accG = accI, accO = accI;
#pragma unroll 1
  for (int k0 = 0; k0 < 480; k0 += 32){
    const float* aseg = (k0 < kD)   ? (hrow + k0)
                      : (k0 < 2*kD) ? (rrow + (k0 - kD))
                                    : (hrow + (k0 - 2*kD));
    v16bf a = loadA_vec(aseg, lane);
    v16bf bI, bF, bG, bO;
    if (k0 < 320){
      bI = loadB_vec(wI + k0, lane); bF = loadB_vec(wF + k0, lane);
      bG = loadB_vec(wG + k0, lane); bO = loadB_vec(wO + k0, lane);
    } else {
      int kl = k0 - 320;
      bI = loadB_vec(uI + kl, lane); bF = loadB_vec(uF + kl, lane);
      bG = loadB_vec(uG + kl, lane); bO = loadB_vec(uO + kl, lane);
    }
    accI = __builtin_amdgcn_wmma_f32_16x16x32_bf16(false, a, false, bI, (short)0, accI, false, false);
    accF = __builtin_amdgcn_wmma_f32_16x16x32_bf16(false, a, false, bF, (short)0, accF, false, false);
    accG = __builtin_amdgcn_wmma_f32_16x16x32_bf16(false, a, false, bG, (short)0, accG, false, false);
    accO = __builtin_amdgcn_wmma_f32_16x16x32_bf16(false, a, false, bO, (short)0, accO, false, false);
  }
  const float bi = bih[0*kD+ncol] + bhh[0*kD+ncol];
  const float bf = bih[1*kD+ncol] + bhh[1*kD+ncol];
  const float bg = bih[2*kD+ncol] + bhh[2*kD+ncol];
  const float bo = bih[3*kD+ncol] + bhh[3*kD+ncol];
#pragma unroll
  for (int rr = 0; rr < 8; ++rr){
    int m = bm + rr + ((lane >> 4) << 3);
    size_t idx = (size_t)m*kD + ncol;
    float iv = sigmoidf_(accI[rr] + bi);
    float fv = sigmoidf_(accF[rr] + bf);
    float gv = tanhf(accG[rr] + bg);
    float ov = sigmoidf_(accO[rr] + bo);
    float cn = fv * c[idx] + iv * gv;
    c[idx] = cn;
    h_out[idx] = ov * tanhf(cn);
  }
}

// ---- elementwise / segment kernels ----
__global__ void fill_kernel(float* __restrict__ p, long long n, float v){
  long long i = (long long)blockIdx.x * blockDim.x + threadIdx.x;
  long long stride = (long long)gridDim.x * blockDim.x;
  for (; i < n; i += stride) p[i] = v;
}

__global__ void count_kernel(const int* __restrict__ seg, float* __restrict__ cnt, int n){
  int i = blockIdx.x * blockDim.x + threadIdx.x;
  if (i < n) atomicAdd(&cnt[seg[i]], 1.f);
}

__global__ void cond_accum_kernel(const int* __restrict__ aidx, const int* __restrict__ gidx,
                                  const float* __restrict__ cond, float* __restrict__ sum,
                                  long long total){
  long long t = (long long)blockIdx.x * blockDim.x + threadIdx.x;
  long long stride = (long long)gridDim.x * blockDim.x;
  for (; t < total; t += stride){
    int i = (int)(t / kD), d = (int)(t % kD);
    atomicAdd(&sum[(size_t)gidx[i]*kD + d], cond[(size_t)aidx[i]*kD + d]);
  }
}

__global__ void cond_div_kernel(float* __restrict__ s, const float* __restrict__ cnt, long long total){
  long long t = (long long)blockIdx.x * blockDim.x + threadIdx.x;
  long long stride = (long long)gridDim.x * blockDim.x;
  for (; t < total; t += stride) s[t] /= fmaxf(cnt[t / kD], 1.f);
}

// e[i] = dot(x[xidx[i]], q[seg[i]]); track per-segment max. One wave per item.
__global__ void s2s_e_kernel(const float* __restrict__ xbase, const int* __restrict__ xidx,
                             const int* __restrict__ seg, const float* __restrict__ q,
                             float* __restrict__ e, float* __restrict__ mbuf, int n){
  int lane = threadIdx.x & 31;
  int i = (int)((blockIdx.x * blockDim.x + threadIdx.x) >> 5);
  if (i >= n) return;
  const float* xr = xbase + (size_t)xidx[i]*kD;
  const float* qr = q + (size_t)seg[i]*kD;
  float s = 0.f;
  for (int d = lane; d < kD; d += 32) s += xr[d] * qr[d];
  s = waveAllReduce(s);
  if (lane == 0){ e[i] = s; atomicMaxF(&mbuf[seg[i]], s); }
}

__global__ void s2s_den_kernel(const float* __restrict__ e, const int* __restrict__ seg,
                               const float* __restrict__ mbuf, float* __restrict__ ea,
                               float* __restrict__ den, int n){
  int i = blockIdx.x * blockDim.x + threadIdx.x;
  if (i >= n) return;
  float v = __expf(e[i] - mbuf[seg[i]]);
  ea[i] = v;
  atomicAdd(&den[seg[i]], v);
}

__global__ void s2s_r_kernel(const float* __restrict__ xbase, const int* __restrict__ xidx,
                             const int* __restrict__ seg, const float* __restrict__ ea,
                             const float* __restrict__ den, float* __restrict__ rbuf, int n){
  int lane = threadIdx.x & 31;
  int i = (int)((blockIdx.x * blockDim.x + threadIdx.x) >> 5);
  if (i >= n) return;
  int s = seg[i];
  float a = ea[i] / den[s];
  const float* xr = xbase + (size_t)xidx[i]*kD;
  float* rr = rbuf + (size_t)s*kD;
  for (int d = lane; d < kD; d += 32) atomicAdd(&rr[d], a * xr[d]);
}

__global__ void mask_kernel(float* __restrict__ h, float* __restrict__ r,
                            const float* __restrict__ cnt, long long total){
  long long t = (long long)blockIdx.x * blockDim.x + threadIdx.x;
  long long stride = (long long)gridDim.x * blockDim.x;
  for (; t < total; t += stride){
    if (cnt[t / kD] <= 0.f){ h[t] = 0.f; r[t] = 0.f; }
  }
}

__global__ void film_kernel(const float* __restrict__ gamma, const float* __restrict__ beta,
                            const float* __restrict__ xf, float* __restrict__ xg, long long total){
  long long t = (long long)blockIdx.x * blockDim.x + threadIdx.x;
  long long stride = (long long)gridDim.x * blockDim.x;
  for (; t < total; t += stride) xg[t] = gamma[t] * xf[t] + beta[t];
}

__global__ void edge_alpha_kernel(const int* __restrict__ src, const int* __restrict__ dst,
                                  const float* __restrict__ msg, const float* __restrict__ att,
                                  float* __restrict__ alpha, float* __restrict__ amax, int n){
  int lane = threadIdx.x & 31;
  int e = (int)((blockIdx.x * blockDim.x + threadIdx.x) >> 5);
  if (e >= n) return;
  const float* md = msg + (size_t)dst[e]*kHALF;
  const float* ms = msg + (size_t)src[e]*kHALF;
  float s = 0.f;
  for (int d = lane; d < kHALF; d += 32) s += (md[d] + ms[d]) * att[d];
  s = waveAllReduce(s);
  if (lane == 0){
    float lr = s > 0.f ? s : 0.2f * s;
    alpha[e] = lr;
    atomicMaxF(&amax[dst[e]], lr);
  }
}

__global__ void edge_den_kernel(const float* __restrict__ alpha, const int* __restrict__ dst,
                                const float* __restrict__ amax, float* __restrict__ ea,
                                float* __restrict__ den, int n){
  int e = blockIdx.x * blockDim.x + threadIdx.x;
  if (e >= n) return;
  float v = __expf(alpha[e] - amax[dst[e]]);
  ea[e] = v;
  atomicAdd(&den[dst[e]], v);
}

__global__ void edge_agg_kernel(const int* __restrict__ src, const int* __restrict__ dst,
                                const float* __restrict__ msg, const float* __restrict__ ea,
                                const float* __restrict__ den, float* __restrict__ agg, int n){
  int lane = threadIdx.x & 31;
  int e = (int)((blockIdx.x * blockDim.x + threadIdx.x) >> 5);
  if (e >= n) return;
  int dd = dst[e];
  float w0 = ea[e] / den[dd];
  const float* ms = msg + (size_t)src[e]*kHALF;
  float* ag = agg + (size_t)dd*kHALF;
  for (int d = lane; d < kHALF; d += 32) atomicAdd(&ag[d], ms[d] * w0);
}

} // namespace

extern "C" void kernel_launch(void* const* d_in, const int* in_sizes, int n_in,
                              void* d_out, int out_size, void* d_ws, size_t ws_size,
                              hipStream_t stream)
{
  (void)in_sizes; (void)n_in; (void)out_size; (void)ws_size;
  const float* x_atom    = (const float*)d_in[0];
  const float* x_group   = (const float*)d_in[1];
  const float* cond_atom = (const float*)d_in[2];
  const float* gw   = (const float*)d_in[3];
  const float* gb   = (const float*)d_in[4];
  const float* aw   = (const float*)d_in[5];
  const float* ab   = (const float*)d_in[6];
  const float* fgW1 = (const float*)d_in[7];
  const float* fgb1 = (const float*)d_in[8];
  const float* fgW2 = (const float*)d_in[9];
  const float* fgb2 = (const float*)d_in[10];
  const float* fbW1 = (const float*)d_in[11];
  const float* fbb1 = (const float*)d_in[12];
  const float* fbW2 = (const float*)d_in[13];
  const float* fbb2 = (const float*)d_in[14];
  const float* aWih = (const float*)d_in[15];
  const float* aWhh = (const float*)d_in[16];
  const float* abih = (const float*)d_in[17];
  const float* abhh = (const float*)d_in[18];
  const float* mw   = (const float*)d_in[19];
  const float* mb   = (const float*)d_in[20];
  const float* gmw  = (const float*)d_in[21];
  const float* gmb  = (const float*)d_in[22];
  const float* gsw  = (const float*)d_in[23];
  const float* gsb  = (const float*)d_in[24];
  const float* gatt = (const float*)d_in[25];
  const float* gWih = (const float*)d_in[26];
  const float* gWhh = (const float*)d_in[27];
  const float* gbih = (const float*)d_in[28];
  const float* gbhh = (const float*)d_in[29];
  const float* pw   = (const float*)d_in[30];
  const float* pb   = (const float*)d_in[31];
  const int* atom_idx  = (const int*)d_in[32];
  const int* group_idx = (const int*)d_in[33];
  const int* eig       = (const int*)d_in[34];
  const int* esrc = eig;
  const int* edst = eig + kE;

  float* out_xa = (float*)d_out;
  float* out_xg = out_xa + (size_t)kNA * kD;

  // ---- scratch layout (bump allocator over d_ws, ~525 MB) ----
  float* wsp = (float*)d_ws;
  size_t off = 0;
  auto alloc = [&](size_t n){ float* p = wsp + off; off += n; return p; };
  float* xa_proj = alloc((size_t)kNA * kD);
  float* hA      = alloc((size_t)kNA * kD);
  float* hB      = alloc((size_t)kNA * kD);
  float* cbuf    = alloc((size_t)kNA * kD);
  float* rbuf    = alloc((size_t)kNA * kD);
  float* ebuf    = alloc((size_t)kNINC);
  float* eabuf   = alloc((size_t)kNINC);
  float* mbuf    = alloc((size_t)kNA);
  float* denbuf  = alloc((size_t)kNA);
  float* cntg    = alloc((size_t)kG);
  float* cnta    = alloc((size_t)kNA);
  float* conds   = alloc((size_t)kG * kD);   // cond_sum -> cond_g in place
  float* xgfrom  = alloc((size_t)kG * kD);
  float* ftmp    = alloc((size_t)kG * kD);
  float* fgamma  = alloc((size_t)kG * kD);
  float* fbeta   = alloc((size_t)kG * kD);
  float* xgbuf   = alloc((size_t)kG * kD);
  float* msg     = alloc((size_t)kG * kHALF);
  float* agg     = alloc((size_t)kG * kHALF);
  float* amax    = alloc((size_t)kG);
  float* den2    = alloc((size_t)kG);
  float* alphab  = alloc((size_t)kE);
  float* ea2     = alloc((size_t)kE);

  auto fill = [&](float* p, long long n, float v){
    int blocks = (int)((n + 1023) / 1024);
    if (blocks > 8192) blocks = 8192;
    if (blocks < 1) blocks = 1;
    fill_kernel<<<blocks, 256, 0, stream>>>(p, n, v);
  };
  auto gemm = [&](const float* A0, const float* A1, int lda0, int lda1, int splitA,
                  const float* W, int ldw, const float* bias,
                  const float* add, int ldadd,
                  float* C, int ldc, int M, int N, int K, int relu){
    int tiles = (M / 16) * (N / 16);
    int blocks = (tiles + 7) / 8;           // 8 waves per 256-thread block
    gemm_bf16_kernel<<<blocks, 256, 0, stream>>>(A0, A1, lda0, lda1, splitA, W, ldw,
                                                 bias, add, ldadd, C, ldc, M, N, K, relu);
  };
  auto gridN = [&](long long n){
    int blocks = (int)((n + 1023) / 1024);
    if (blocks > 8192) blocks = 8192;
    if (blocks < 1) blocks = 1;
    return blocks;
  };

  float* hin = hA;
  float* hout = hB;
  auto s2s = [&](int R, int NI, const float* xbase, const int* xidx, const int* seg,
                 const float* Wih, const float* Whh, const float* bih, const float* bhh){
    fill(hin, (long long)R * kD, 0.f);
    fill(cbuf, (long long)R * kD, 0.f);
    fill(rbuf, (long long)R * kD, 0.f);
    for (int step = 0; step < 2; ++step){
      int tiles = (R / 16) * (kD / 16);
      lstm_gemm_cell_kernel<<<(tiles + 7) / 8, 256, 0, stream>>>(hin, rbuf, Wih, Whh,
                                                                 bih, bhh, cbuf, hout, R);
      fill(mbuf, R, -INFINITY);
      fill(denbuf, R, 0.f);
      fill(rbuf, (long long)R * kD, 0.f);
      s2s_e_kernel<<<(NI + 7) / 8, 256, 0, stream>>>(xbase, xidx, seg, hout, ebuf, mbuf, NI);
      s2s_den_kernel<<<(NI + 255) / 256, 256, 0, stream>>>(ebuf, seg, mbuf, eabuf, denbuf, NI);
      s2s_r_kernel<<<(NI + 7) / 8, 256, 0, stream>>>(xbase, xidx, seg, eabuf, denbuf, rbuf, NI);
      float* t = hin; hin = hout; hout = t;   // final q ends in hin
    }
  };

  // ================= Phase A: atoms -> groups =================
  fill(cntg, kG, 0.f);
  count_kernel<<<(kNINC + 255) / 256, 256, 0, stream>>>(group_idx, cntg, kNINC);
  fill(conds, (long long)kG * kD, 0.f);
  cond_accum_kernel<<<gridN((long long)kNINC * kD), 256, 0, stream>>>(
      atom_idx, group_idx, cond_atom, conds, (long long)kNINC * kD);
  cond_div_kernel<<<gridN((long long)kG * kD), 256, 0, stream>>>(conds, cntg, (long long)kG * kD);

  // xa_proj = x_atom @ aw.T + ab
  gemm(x_atom, x_atom, kD, kD, kD, aw, kD, ab, nullptr, 0, xa_proj, kD, kNA, kD, kD, 0);

  // Set2Set atoms->groups (x = xa_proj[atom_idx], seg = group_idx)
  s2s(kG, kNINC, xa_proj, atom_idx, group_idx, aWih, aWhh, abih, abhh);
  mask_kernel<<<gridN((long long)kG * kD), 256, 0, stream>>>(hin, rbuf, cntg, (long long)kG * kD);

  // xg_from_atom = [q|r] @ mw.T + mb
  gemm(hin, rbuf, kD, kD, kD, mw, 2 * kD, mb, nullptr, 0, xgfrom, kD, kG, kD, 2 * kD, 0);

  // FiLM gamma/beta MLPs on cond_g
  gemm(conds, conds, kD, kD, kD, fgW1, kD, fgb1, nullptr, 0, ftmp, kD, kG, kD, kD, 1);
  gemm(ftmp, ftmp, kD, kD, kD, fgW2, kD, fgb2, nullptr, 0, fgamma, kD, kG, kD, kD, 0);
  gemm(conds, conds, kD, kD, kD, fbW1, kD, fbb1, nullptr, 0, ftmp, kD, kG, kD, kD, 1);
  gemm(ftmp, ftmp, kD, kD, kD, fbW2, kD, fbb2, nullptr, 0, fbeta, kD, kG, kD, kD, 0);
  film_kernel<<<gridN((long long)kG * kD), 256, 0, stream>>>(fgamma, fbeta, xgfrom, xgbuf,
                                                             (long long)kG * kD);

  // xg0 -> out_xg columns [0,80)  (K=40, uses the guarded tail tile once)
  gemm(x_group, x_group, 40, 40, 40, gw, 40, gb, nullptr, 0, out_xg, kD, kG, kHALF, 40, 0);

  // GCN: msg, edge attention softmax, aggregate, self + relu -> out_xg cols [80,160)
  gemm(xgbuf, xgbuf, kD, kD, kD, gmw, kD, gmb, nullptr, 0, msg, kHALF, kG, kHALF, kD, 0);
  fill(amax, kG, -INFINITY);
  fill(den2, kG, 0.f);
  fill(agg, (long long)kG * kHALF, 0.f);
  edge_alpha_kernel<<<(kE + 7) / 8, 256, 0, stream>>>(esrc, edst, msg, gatt, alphab, amax, kE);
  edge_den_kernel<<<(kE + 255) / 256, 256, 0, stream>>>(alphab, edst, amax, ea2, den2, kE);
  edge_agg_kernel<<<(kE + 7) / 8, 256, 0, stream>>>(esrc, edst, msg, ea2, den2, agg, kE);
  gemm(xgbuf, xgbuf, kD, kD, kD, gsw, kD, gsb, agg, kHALF, out_xg + kHALF, kD, kG, kHALF, kD, 1);

  // ================= Phase B: groups -> atoms =================
  fill(cnta, kNA, 0.f);
  count_kernel<<<(kNINC + 255) / 256, 256, 0, stream>>>(atom_idx, cnta, kNINC);

  // Set2Set groups->atoms (x = out_xg[group_idx], seg = atom_idx)
  s2s(kNA, kNINC, out_xg, group_idx, atom_idx, gWih, gWhh, gbih, gbhh);
  mask_kernel<<<gridN((long long)kNA * kD), 256, 0, stream>>>(hin, rbuf, cnta, (long long)kNA * kD);

  // xa_out = x_atom + [q|r] @ pw.T + pb
  gemm(hin, rbuf, kD, kD, kD, pw, 2 * kD, pb, x_atom, kD, out_xa, kD, kNA, kD, 2 * kD, 0);
}
